// EncoderLayer_91250875171634
// MI455X (gfx1250) — matrix operand order repaired and checked by reading, not compiled, not run on hardware
//
#include <hip/hip_runtime.h>
#include <hip/hip_bf16.h>

typedef __bf16 bf16;
typedef __attribute__((ext_vector_type(16))) __bf16 v16bf;
typedef __attribute__((ext_vector_type(8)))  __bf16 v8bf;
typedef __attribute__((ext_vector_type(8)))  float  v8f;

#define BB  2
#define SS  4096
#define DM  256
#define HH  8
#define HKC 2048   /* H*KD */
#define DFF 1024
#define MM  (BB*SS)

#define USE_ASYNC_LDS 1

// ---------- helpers ----------
__device__ __forceinline__ bf16 f2bf(float f) {
  union { float f; unsigned u; } c; c.f = f;
  unsigned r = (c.u + 0x7FFFu + ((c.u >> 16) & 1u)) >> 16;
  unsigned short us = (unsigned short)r;
  bf16 b; __builtin_memcpy(&b, &us, 2); return b;
}

// 16-byte global -> LDS copy on the CDNA5 ASYNCcnt path (generic LDS ptr bits[31:0]
// are the LDS byte offset per ISA 10.2 aperture rules).
__device__ __forceinline__ void cp16_lds(const bf16* g, bf16* l) {
#if USE_ASYNC_LDS
  unsigned loff = (unsigned)(uintptr_t)l;
  asm volatile("global_load_async_to_lds_b128 %0, %1, off"
               :: "v"(loff), "v"(g) : "memory");
#else
  *(v8bf*)l = *(const v8bf*)g;
#endif
}
__device__ __forceinline__ void wait_cp() {
#if USE_ASYNC_LDS
  asm volatile("s_wait_asynccnt 0" ::: "memory");
#endif
}

// A-fragment: lane covers row (lane&15); K = 16h..16h+15
__device__ __forceinline__ v16bf frag_runs(const bf16* p) {
  v8bf lo = *(const v8bf*)p;
  v8bf hi = *(const v8bf*)(p + 8);
  return __builtin_shufflevector(lo, hi, 0,1,2,3,4,5,6,7,8,9,10,11,12,13,14,15);
}
// B-fragment from N-major storage: lane covers col (lane&15); K = 16h..16h+15
__device__ __forceinline__ v16bf frag_contig(const bf16* p) {
  v8bf lo = *(const v8bf*)p;
  v8bf hi = *(const v8bf*)(p + 8);
  return __builtin_shufflevector(lo, hi, 0,1,2,3,4,5,6,7,8,9,10,11,12,13,14,15);
}
__device__ __forceinline__ v8f wmma16(v16bf a, v16bf b, v8f c) {
  return __builtin_amdgcn_wmma_f32_16x16x32_bf16(false, a, false, b, (short)0, c, false, false);
}

// ---------- fp32 -> bf16 packers ----------
__global__ void cvt_bf16(const float* __restrict__ s, bf16* __restrict__ d, int n) {
  int i = blockIdx.x * blockDim.x + threadIdx.x;
  if (i < n) d[i] = f2bf(s[i]);
}
// s is [R][C] row-major; d becomes [C][R] (N-major weights for GEMM B staging)
__global__ void cvt_bf16_t(const float* __restrict__ s, bf16* __restrict__ d, int R, int C) {
  int i = blockIdx.x * blockDim.x + threadIdx.x;
  if (i < R * C) {
    int r = i / C, c = i - r * C;
    d[(size_t)c * R + r] = f2bf(s[i]);
  }
}

// ---------- WMMA GEMM: C[M,N] = act(A[M,K] x Bt[N,K]^T + bias) ----------
// 128x128 block tile, 8 waves; each wave: 16 rows x 128 cols (8 WMMA/k-step).
// Double-buffered async global->LDS staging, one barrier per k-step.
__global__ __launch_bounds__(256)
void gemm_wmma(const bf16* __restrict__ A, const bf16* __restrict__ Bt,
               const float* __restrict__ bias, float* __restrict__ Cf,
               bf16* __restrict__ Cb, int M, int N, int K, int act) {
  __shared__ bf16 As[2][128][40];   // pitch 40 elems = 80B (16B multiple)
  __shared__ bf16 Bs[2][128][40];   // Bt tile: [n][k]
  const int tid = threadIdx.x;
  const int wv = tid >> 5, lane = tid & 31;
  const int nn = lane & 15, hf = lane >> 4;
  const int m0 = blockIdx.y * 128, n0 = blockIdx.x * 128;
  const int row = tid >> 1, seg = tid & 1;     // 128 rows x 2 segs of 16 elems
  v8f acc[8] = {};
  const bf16* gA = A  + (size_t)(m0 + row) * K + seg * 16;
  const bf16* gB = Bt + (size_t)(n0 + row) * K + seg * 16;

  // prologue: stage k-step 0 into buffer 0
  cp16_lds(gA,     &As[0][row][seg*16]);
  cp16_lds(gA + 8, &As[0][row][seg*16 + 8]);
  cp16_lds(gB,     &Bs[0][row][seg*16]);
  cp16_lds(gB + 8, &Bs[0][row][seg*16 + 8]);
  wait_cp();
  __syncthreads();

  const int nsteps = K >> 5;
  for (int s = 0; s < nsteps; ++s) {
    const int cur = s & 1;
    if (s + 1 < nsteps) {           // async-stage next tile while computing
      const bf16* nA = gA + (s + 1) * 32;
      const bf16* nB = gB + (s + 1) * 32;
      cp16_lds(nA,     &As[cur ^ 1][row][seg*16]);
      cp16_lds(nA + 8, &As[cur ^ 1][row][seg*16 + 8]);
      cp16_lds(nB,     &Bs[cur ^ 1][row][seg*16]);
      cp16_lds(nB + 8, &Bs[cur ^ 1][row][seg*16 + 8]);
    }
    // batch all fragment loads, then issue the 8 WMMAs back-to-back
    v16bf af = frag_runs(&As[cur][wv*16 + nn][hf*16]);
    v16bf bfr[8];
    #pragma unroll
    for (int nt = 0; nt < 8; ++nt)
      bfr[nt] = frag_contig(&Bs[cur][nt*16 + nn][hf*16]);
    #pragma unroll
    for (int nt = 0; nt < 8; ++nt)
      acc[nt] = wmma16(af, bfr[nt], acc[nt]);
    wait_cp();
    __syncthreads();
  }

  #pragma unroll
  for (int nt = 0; nt < 8; ++nt) {
    #pragma unroll
    for (int r = 0; r < 8; ++r) {
      int m = m0 + wv*16 + r + 8*hf;
      int n = n0 + nt*16 + nn;
      float v = acc[nt][r];
      if (bias) v += bias[n];
      if (act)  v = v > 0.f ? v : 0.f;
      if (Cf) Cf[(size_t)m * N + n] = v;
      if (Cb) Cb[(size_t)m * N + n] = f2bf(v);
    }
  }
}

// ---------- sliding-window attention: one block per (b,h,q-tile of 16) ----------
__global__ __launch_bounds__(128)
void attn_win(const bf16* __restrict__ Q, const bf16* __restrict__ Kmat,
              const bf16* __restrict__ V, bf16* __restrict__ Ctx,
              const int* __restrict__ w2p) {
  __shared__ bf16 Qs[16][264];    // 16x256 Q tile
  __shared__ float Sc[16][280];   // scores, up to 272 keys
  __shared__ bf16 Pr[16][296];    // probs, zero-padded to 288 (9 k32 chunks)
  __shared__ bf16 Vt[256][40];    // V chunk transposed: [feature][key in chunk]
  const int tid = threadIdx.x, wv = tid >> 5, lane = tid & 31;
  const int nn = lane & 15, hf = lane >> 4;
  const int q0 = blockIdx.x * 16;
  const int b = blockIdx.y >> 3, h = blockIdx.y & 7;
  int w = w2p[0]; if (w > 128) w = 128;
  const int lo = q0 - w;
  const int nk = 16 + 2 * w;                 // 272 for w=128
  const int ntile = (nk + 15) >> 4;          // 17
  const bf16* Qp = Q    + (size_t)b * SS * HKC + (size_t)h * DM;
  const bf16* Kp = Kmat + (size_t)b * SS * HKC + (size_t)h * DM;
  const bf16* Vp = V    + (size_t)b * SS * HKC + (size_t)h * DM;
  { // load Q tile into LDS (reused ~17x) via async path
    int r = tid >> 3, seg = tid & 7;
    const bf16* gp = Qp + (size_t)(q0 + r) * HKC + seg * 32;
    cp16_lds(gp,      &Qs[r][seg*32]);
    cp16_lds(gp + 8,  &Qs[r][seg*32 + 8]);
    cp16_lds(gp + 16, &Qs[r][seg*32 + 16]);
    cp16_lds(gp + 24, &Qs[r][seg*32 + 24]);
  }
  wait_cp();
  __syncthreads();
  // ---- phase 1: scores = Q K^T / 16 ; K fragments streamed straight from global.
  for (int kt = wv; kt < ntile; kt += 4) {
    int key = lo + kt*16 + nn;
    int keyc = key < 0 ? 0 : (key > SS-1 ? SS-1 : key);   // clamped rows get masked later
    const bf16* bp = Kp + (size_t)keyc * HKC + hf * 16;
    v16bf bfr[8];
    #pragma unroll
    for (int c = 0; c < 8; ++c)                   // hoist all global loads
      bfr[c] = frag_contig(bp + c*32);
    v8f acc = {};
    #pragma unroll
    for (int c = 0; c < 8; ++c) {                 // KD=256 = 8 x k32
      v16bf af = frag_runs(&Qs[nn][c*32 + hf*16]);
      acc = wmma16(af, bfr[c], acc);
    }
    #pragma unroll
    for (int r = 0; r < 8; ++r)
      Sc[r + 8*hf][kt*16 + nn] = acc[r] * 0.0625f;  // 1/sqrt(256)
  }
  __syncthreads();
  // ---- phase 2: masked softmax (8 lanes per query row)
  {
    int r = tid >> 3, g = tid & 7;
    int gq = q0 + r;
    int klo = gq - w; if (klo < 0) klo = 0;
    int khi = gq + w; if (khi > SS-1) khi = SS-1;
    float m = -3.4e38f;
    for (int j = g; j < nk; j += 8) {
      int k = lo + j;
      if (k >= klo && k <= khi) { float s = Sc[r][j]; m = m > s ? m : s; }
    }
    for (int o = 1; o < 8; o <<= 1) { float t = __shfl_xor(m, o, 32); m = m > t ? m : t; }
    float sum = 0.f;
    for (int j = g; j < nk; j += 8) {
      int k = lo + j;
      float e = 0.f;
      if (k >= klo && k <= khi) e = __expf(Sc[r][j] - m);
      Sc[r][j] = e;
      sum += e;
    }
    for (int o = 1; o < 8; o <<= 1) sum += __shfl_xor(sum, o, 32);
    float inv = 1.f / sum;
    for (int j = g; j < 288; j += 8)
      Pr[r][j] = f2bf((j < nk) ? Sc[r][j] * inv : 0.f);
  }
  __syncthreads();
  // ---- phase 3: ctx = P V ; stage V chunks transposed through LDS
  v8f acc4[4] = {};
  const int nchunk = (nk + 31) >> 5;              // 9
  for (int c = 0; c < nchunk; ++c) {
    __syncthreads();
    { // 32 keys x 256 features -> Vt[feature][key]
      int kl = tid >> 2, fs = (tid & 3) * 64;
      int key = lo + c*32 + kl;
      int keyc = key < 0 ? 0 : (key > SS-1 ? SS-1 : key);
      const bf16* gp = Vp + (size_t)keyc * HKC + fs;
      #pragma unroll
      for (int i = 0; i < 8; ++i) {
        v8bf x = *(const v8bf*)(gp + i * 8);
        #pragma unroll
        for (int j = 0; j < 8; ++j) Vt[fs + i*8 + j][kl] = x[j];
      }
    }
    __syncthreads();
    v16bf af = frag_runs(&Pr[nn][c*32 + hf*16]);
    v16bf bfr[4];
    #pragma unroll
    for (int nt = 0; nt < 4; ++nt)
      bfr[nt] = frag_contig(&Vt[(wv*4 + nt)*16 + nn][hf*16]);
    #pragma unroll
    for (int nt = 0; nt < 4; ++nt)                 // wave owns 64 features
      acc4[nt] = wmma16(af, bfr[nt], acc4[nt]);
  }
  #pragma unroll
  for (int nt = 0; nt < 4; ++nt) {
    #pragma unroll
    for (int r = 0; r < 8; ++r) {
      int m = q0 + r + 8*hf;
      int f = (wv*4 + nt)*16 + nn;
      Ctx[((size_t)b*SS + m) * HKC + h*DM + f] = f2bf(acc4[nt][r]);
    }
  }
}

// ---------- fused residual + LayerNorm ----------
__global__ __launch_bounds__(256)
void resid_ln(const float* __restrict__ xin, const float* __restrict__ yin,
              const float* __restrict__ gam, const float* __restrict__ bet,
              float* __restrict__ outF, bf16* __restrict__ outB) {
  __shared__ float red[8];
  const int t = threadIdx.x;
  const size_t row = blockIdx.x;
  float v = xin[row*DM + t] + yin[row*DM + t];
  float s = v;
  for (int o = 16; o; o >>= 1) s += __shfl_xor(s, o, 32);
  if ((t & 31) == 0) red[t >> 5] = s;
  __syncthreads();
  float mu = 0.f;
  #pragma unroll
  for (int i = 0; i < 8; ++i) mu += red[i];
  mu *= (1.f / DM);
  __syncthreads();
  float d = v - mu;
  float s2 = d * d;
  for (int o = 16; o; o >>= 1) s2 += __shfl_xor(s2, o, 32);
  if ((t & 31) == 0) red[t >> 5] = s2;
  __syncthreads();
  float var = 0.f;
  #pragma unroll
  for (int i = 0; i < 8; ++i) var += red[i];
  var *= (1.f / DM);
  float rs = rsqrtf(var + 1e-3f);
  float res = d * rs * gam[t] + bet[t];
  outF[row*DM + t] = res;
  if (outB) outB[row*DM + t] = f2bf(res);
}

// ---------- host launcher ----------
extern "C" void kernel_launch(void* const* d_in, const int* in_sizes, int n_in,
                              void* d_out, int out_size, void* d_ws, size_t ws_size,
                              hipStream_t stream) {
  const float* x   = (const float*)d_in[0];
  const int*   w2  = (const int*)  d_in[1];
  const float* Wq  = (const float*)d_in[2];
  const float* bq  = (const float*)d_in[3];
  const float* Wk  = (const float*)d_in[4];
  const float* bk  = (const float*)d_in[5];
  const float* Wv  = (const float*)d_in[6];
  const float* bv  = (const float*)d_in[7];
  const float* Wo  = (const float*)d_in[8];
  const float* bo  = (const float*)d_in[9];
  const float* W1  = (const float*)d_in[10];
  const float* b1  = (const float*)d_in[11];
  const float* W2m = (const float*)d_in[12];
  const float* b2  = (const float*)d_in[13];
  const float* g1  = (const float*)d_in[14];
  const float* be1 = (const float*)d_in[15];
  const float* g2  = (const float*)d_in[16];
  const float* be2 = (const float*)d_in[17];

  char* ws = (char*)d_ws;
  size_t off = 0;
  auto carve = [&](size_t bytes) -> void* {
    void* p = ws + off;
    off += (bytes + 255) & ~(size_t)255;
    return p;
  };
  bf16* xb   = (bf16*)carve((size_t)MM * DM * 2);
  bf16* wqbT = (bf16*)carve((size_t)DM * HKC * 2);   // [HKC][DM]
  bf16* wkbT = (bf16*)carve((size_t)DM * HKC * 2);
  bf16* wvbT = (bf16*)carve((size_t)DM * HKC * 2);
  bf16* wobT = (bf16*)carve((size_t)HKC * DM * 2);   // [DM][HKC]
  bf16* w1bT = (bf16*)carve((size_t)DM * DFF * 2);   // [DFF][DM]
  bf16* w2bT = (bf16*)carve((size_t)DFF * DM * 2);   // [DM][DFF]
  bf16* qb   = (bf16*)carve((size_t)MM * HKC * 2);
  bf16* kb   = (bf16*)carve((size_t)MM * HKC * 2);
  bf16* vb   = (bf16*)carve((size_t)MM * HKC * 2);
  bf16* ctxb = (bf16*)carve((size_t)MM * HKC * 2);
  float* tmp1 = (float*)carve((size_t)MM * DM * 4);
  float* y1   = (float*)carve((size_t)MM * DM * 4);
  bf16* y1b   = (bf16*)carve((size_t)MM * DM * 2);
  bf16* hb    = qb;      // alias: q dead after attention (16 MB < 32 MB)
  float* tmp2 = tmp1;    // alias: tmp1 dead after first LN

  cvt_bf16<<<(MM*DM + 255)/256, 256, 0, stream>>>(x, xb, MM * DM);
  // weights pre-transposed to N-major so GEMM B staging is contiguous/async-able
  cvt_bf16_t<<<(DM*HKC + 255)/256, 256, 0, stream>>>(Wq,  wqbT, DM,  HKC);
  cvt_bf16_t<<<(DM*HKC + 255)/256, 256, 0, stream>>>(Wk,  wkbT, DM,  HKC);
  cvt_bf16_t<<<(DM*HKC + 255)/256, 256, 0, stream>>>(Wv,  wvbT, DM,  HKC);
  cvt_bf16_t<<<(HKC*DM + 255)/256, 256, 0, stream>>>(Wo,  wobT, HKC, DM);
  cvt_bf16_t<<<(DM*DFF + 255)/256, 256, 0, stream>>>(W1,  w1bT, DM,  DFF);
  cvt_bf16_t<<<(DFF*DM + 255)/256, 256, 0, stream>>>(W2m, w2bT, DFF, DM);

  dim3 blk(256);
  dim3 gqkv(HKC/128, MM/128);
  gemm_wmma<<<gqkv, blk, 0, stream>>>(xb, wqbT, bq, nullptr, qb, MM, HKC, DM, 0);
  gemm_wmma<<<gqkv, blk, 0, stream>>>(xb, wkbT, bk, nullptr, kb, MM, HKC, DM, 0);
  gemm_wmma<<<gqkv, blk, 0, stream>>>(xb, wvbT, bv, nullptr, vb, MM, HKC, DM, 0);
  attn_win<<<dim3(SS/16, BB*HH), dim3(128), 0, stream>>>(qb, kb, vb, ctxb, w2);
  gemm_wmma<<<dim3(DM/128, MM/128), blk, 0, stream>>>(ctxb, wobT, bo, tmp1, nullptr, MM, DM, HKC, 0);
  resid_ln<<<MM, 256, 0, stream>>>(x, tmp1, g1, be1, y1, y1b);
  gemm_wmma<<<dim3(DFF/128, MM/128), blk, 0, stream>>>(y1b, w1bT, b1, nullptr, hb, MM, DFF, DM, 1);
  gemm_wmma<<<dim3(DM/128, MM/128), blk, 0, stream>>>(hb, w2bT, b2, tmp2, nullptr, MM, DM, DFF, 0);
  resid_ln<<<MM, 256, 0, stream>>>(y1, tmp2, g2, be2, (float*)d_out, nullptr);
}
